// RnnSearchModule_18116172054987
// MI455X (gfx1250) — matrix operand order repaired and checked by reading, not compiled, not run on hardware
//
#include <hip/hip_runtime.h>

// ---------------------------------------------------------------------------
// ENAS RNN cell, T=512 steps, B=128, C=512.
//   step: n0 = tanh(x@Wx + h@Wh); n1 = relu(n0@W1);
//         out = 0.5*(sigmoid(n1@W2) + tanh(n1@W3)); h = out
//
// MI455X (gfx1250) strategy:
//  * Xp = X @ Wx precomputed for all T as one big parallel bf16 WMMA GEMM,
//    written directly into d_out (each slot consumed at step t, then
//    overwritten with the real output).
//  * KEY: the recurrence never mixes batch rows -> B=128 splits into 8
//    independent 16-row stripes. Each stripe's full T=512 scan runs in ONE
//    persistent 1024-thread workgroup (32 waves = 32 N-tiles of 16x16),
//    with h/n0/n1 stripes resident in LDS and workgroup split-barriers
//    (s_barrier_signal/_wait) between the 3 dependent GEMM stages.
//    This removes ~1536 serial kernel launches entirely.
//  * Weights pre-swizzled into exact 16x16x32 bf16 B-fragment order so each
//    wave's B load is a contiguous 1KB L2-resident block.
//  * LDS row stride padded to 520 bf16 (1040B = 260 dwords, 260 % 64 == 4)
//    so A-fragment ds_load_b128 across lanes 0-15 is bank-conflict-free.
// ---------------------------------------------------------------------------

#define Tn   512
#define Bn   128
#define Cn   512
#define KC   (Cn / 32)    // 16 k-chunks of 32
#define LDP  520          // padded LDS row stride (bf16 elements)

typedef __attribute__((ext_vector_type(16))) __bf16 v16bf;
typedef __attribute__((ext_vector_type(8)))  __bf16 v8bf;
typedef __attribute__((ext_vector_type(8)))  float  v8f;

#define WMMA_BF16(a, b, c) \
    __builtin_amdgcn_wmma_f32_16x16x32_bf16(false, (a), false, (b), (short)0, (c), false, false)

// ---- fragment loaders ------------------------------------------------------

// A fragment (16x32 bf16) from an LDS stripe [16][LDP].
// ISA layout: lanes 0-15 row M=lane, K = kb + {0..7, 16..23};
//             lanes 16-31 row M=lane-16, K = kb + {8..15, 24..31}.
__device__ __forceinline__ v16bf lds_a_frag(const __bf16* S, int kb, int lane) {
    const int half = lane >> 4;
    const int lrow = lane & 15;
    const __bf16* p = S + lrow * LDP + kb + half * 8;
    union { v16bf v; v8bf h[2]; } u;
    u.h[0] = *(const v8bf*)(p);
    u.h[1] = *(const v8bf*)(p + 16);
    return u.v;
}

// B fragment from pre-swizzled weights: contiguous 1KB per (ntile,kchunk):
// Ws[(((nt*KC + kc)*32) + lane)*16 + e]  holds  W[k][n],
//   n = nt*16 + (lane&15),  k = kc*32 + (e>>3)*16 + (lane>>4)*8 + (e&7).
__device__ __forceinline__ v16bf load_b_frag(const __bf16* Ws, int nt, int kc, int lane) {
    const __bf16* p = Ws + ((((size_t)nt * KC + kc) * 32 + lane) << 4);
    union { v16bf v; v8bf h[2]; } u;
    u.h[0] = *(const v8bf*)(p);
    u.h[1] = *(const v8bf*)(p + 8);
    return u.v;
}

// ---- activations -----------------------------------------------------------

__device__ __forceinline__ float fast_tanh(float x) {
    x = fminf(fmaxf(x, -15.f), 15.f);
    float e = __expf(2.f * x);
    return (e - 1.f) / (e + 1.f);
}
__device__ __forceinline__ float fast_sigmoid(float x) {
    x = fminf(fmaxf(x, -30.f), 30.f);
    return 1.f / (1.f + __expf(-x));
}

// ---- one-time weight swizzle ----------------------------------------------

__global__ void prep_w_kernel(const float* __restrict__ W, __bf16* __restrict__ Ws) {
    int idx  = blockIdx.x * 256 + threadIdx.x;     // 0 .. 512*512-1
    int e    = idx & 15;
    int lane = (idx >> 4) & 31;
    int kc   = (idx >> 9) & (KC - 1);
    int nt   = idx >> 13;                          // 0..31
    int n = nt * 16 + (lane & 15);
    int k = kc * 32 + (e >> 3) * 16 + (lane >> 4) * 8 + (e & 7);
    Ws[idx] = (__bf16)W[(size_t)k * Cn + n];
}

// ---- big parallel GEMM: Xp = X @ Wx  (fp32 in, bf16 WMMA, fp32 out) --------
// M = T*B = 65536, N = K = 512. Wave tile = 16M x 64N (4 accumulators).
__global__ void xwx_kernel(const float* __restrict__ X,
                           const __bf16* __restrict__ Wxs,
                           float* __restrict__ Xp) {
    const int lane = threadIdx.x & 31;
    const int wid  = (blockIdx.x * 256 + threadIdx.x) >> 5;   // 0..32767
    const int mt   = wid >> 3;                                // 0..4095
    const int ng   = wid & 7;                                 // group of 4 n-tiles
    const int m0   = mt * 16;
    const int half = lane >> 4;
    const int lrow = lane & 15;

    v8f acc0 = {}, acc1 = {}, acc2 = {}, acc3 = {};

    for (int kc = 0; kc < KC; ++kc) {
        const int kb = kc * 32;
        const float* ar = X + (size_t)(m0 + lrow) * Cn + kb + half * 8;
        v16bf a;
#pragma unroll
        for (int i = 0; i < 8; ++i) a[i] = (__bf16)ar[i];
#pragma unroll
        for (int i = 0; i < 8; ++i) a[8 + i] = (__bf16)ar[16 + i];

        v16bf b0 = load_b_frag(Wxs, ng * 4 + 0, kc, lane);
        v16bf b1 = load_b_frag(Wxs, ng * 4 + 1, kc, lane);
        v16bf b2 = load_b_frag(Wxs, ng * 4 + 2, kc, lane);
        v16bf b3 = load_b_frag(Wxs, ng * 4 + 3, kc, lane);

        acc0 = WMMA_BF16(a, b0, acc0);
        acc1 = WMMA_BF16(a, b1, acc1);
        acc2 = WMMA_BF16(a, b2, acc2);
        acc3 = WMMA_BF16(a, b3, acc3);
    }

#pragma unroll
    for (int r = 0; r < 8; ++r) {
        const size_t row = (size_t)(m0 + r + half * 8) * Cn;
        Xp[row + (ng * 4 + 0) * 16 + lrow] = acc0[r];
        Xp[row + (ng * 4 + 1) * 16 + lrow] = acc1[r];
        Xp[row + (ng * 4 + 2) * 16 + lrow] = acc2[r];
        Xp[row + (ng * 4 + 3) * 16 + lrow] = acc3[r];
    }
}

// ---- persistent per-stripe scan kernel -------------------------------------
// grid = 8 blocks (one 16-row batch stripe each), 1024 threads = 32 waves.
// Wave nt owns output columns [nt*16, nt*16+16) of every stage.
__global__ __launch_bounds__(1024) void scan_kernel(
        const float*  __restrict__ S0,       // [B,C] initial state (fp32)
        const __bf16* __restrict__ Whs,
        const __bf16* __restrict__ W1s,
        const __bf16* __restrict__ W2s,
        const __bf16* __restrict__ W3s,
        float* __restrict__ io)              // d_out: Xp in, outputs out
{
    __shared__ __align__(16) __bf16 hS [16 * LDP];
    __shared__ __align__(16) __bf16 n0S[16 * LDP];
    __shared__ __align__(16) __bf16 n1S[16 * LDP];

    const int tid  = threadIdx.x;
    const int lane = tid & 31;
    const int nt   = tid >> 5;              // 0..31 : N-tile of this wave
    const int m0   = blockIdx.x * 16;       // global batch-row base
    const int half = lane >> 4;
    const int lrow = lane & 15;
    const int col  = nt * 16 + lrow;

    // init h stripe from initial state
    for (int i = tid; i < 16 * Cn; i += 1024) {
        int r = i >> 9, c = i & (Cn - 1);
        hS[r * LDP + c] = (__bf16)S0[(size_t)(m0 + r) * Cn + c];
    }
    __syncthreads();

    for (int t = 0; t < Tn; ++t) {
        float* slab = io + (size_t)t * Bn * Cn;   // Xp_t on entry, out_t on exit

        // ---- stage 1: n0 = tanh(Xp_t + h @ Wh) ----
        v8f acc;
#pragma unroll
        for (int r = 0; r < 8; ++r)
            acc[r] = slab[(size_t)(m0 + r + half * 8) * Cn + col];

#pragma unroll 4
        for (int kc = 0; kc < KC; ++kc) {
            if (kc + 1 < KC)
                __builtin_prefetch(Whs + ((((size_t)nt * KC + kc + 1) * 32 + lane) << 4), 0, 0);
            v16bf a = lds_a_frag(hS, kc * 32, lane);
            v16bf b = load_b_frag(Whs, nt, kc, lane);
            acc = WMMA_BF16(a, b, acc);
        }
#pragma unroll
        for (int r = 0; r < 8; ++r)
            n0S[(r + half * 8) * LDP + col] = (__bf16)fast_tanh(acc[r]);
        __syncthreads();

        // ---- stage 2: n1 = relu(n0 @ W1) ----
        v8f acc1 = {};
#pragma unroll 4
        for (int kc = 0; kc < KC; ++kc) {
            if (kc + 1 < KC)
                __builtin_prefetch(W1s + ((((size_t)nt * KC + kc + 1) * 32 + lane) << 4), 0, 0);
            v16bf a = lds_a_frag(n0S, kc * 32, lane);
            v16bf b = load_b_frag(W1s, nt, kc, lane);
            acc1 = WMMA_BF16(a, b, acc1);
        }
#pragma unroll
        for (int r = 0; r < 8; ++r) {
            float v = acc1[r];
            n1S[(r + half * 8) * LDP + col] = (__bf16)(v > 0.f ? v : 0.f);
        }
        __syncthreads();

        // ---- stage 3: out = 0.5*(sigmoid(n1@W2) + tanh(n1@W3)); h = out ----
        v8f acc2 = {}, acc3 = {};
#pragma unroll 4
        for (int kc = 0; kc < KC; ++kc) {
            v16bf a  = lds_a_frag(n1S, kc * 32, lane);
            v16bf b2 = load_b_frag(W2s, nt, kc, lane);
            v16bf b3 = load_b_frag(W3s, nt, kc, lane);
            acc2 = WMMA_BF16(a, b2, acc2);
            acc3 = WMMA_BF16(a, b3, acc3);
        }
#pragma unroll
        for (int r = 0; r < 8; ++r) {
            float o = 0.5f * (fast_sigmoid(acc2[r]) + fast_tanh(acc3[r]));
            slab[(size_t)(m0 + r + half * 8) * Cn + col] = o;   // output_t
            hS[(r + half * 8) * LDP + col] = (__bf16)o;         // next state
        }
        __syncthreads();
    }
}

// ---------------------------------------------------------------------------

extern "C" void kernel_launch(void* const* d_in, const int* in_sizes, int n_in,
                              void* d_out, int out_size, void* d_ws, size_t ws_size,
                              hipStream_t stream) {
    const float* X  = (const float*)d_in[0];   // [T,B,C]
    const float* S0 = (const float*)d_in[1];   // [1,B,C]
    const float* Wx = (const float*)d_in[2];
    const float* Wh = (const float*)d_in[3];
    const float* W1 = (const float*)d_in[4];
    const float* W2 = (const float*)d_in[5];
    const float* W3 = (const float*)d_in[6];
    float* out = (float*)d_out;

    const size_t WSZ = (size_t)Cn * Cn;        // elements per swizzled weight
    __bf16* Wxs = (__bf16*)d_ws;
    __bf16* Whs = Wxs + WSZ;
    __bf16* W1s = Whs + WSZ;
    __bf16* W2s = W1s + WSZ;
    __bf16* W3s = W2s + WSZ;

    // one-time weight swizzle (2.5 MB bf16 total, L2-resident thereafter)
    prep_w_kernel<<<1024, 256, 0, stream>>>(Wx, Wxs);
    prep_w_kernel<<<1024, 256, 0, stream>>>(Wh, Whs);
    prep_w_kernel<<<1024, 256, 0, stream>>>(W1, W1s);
    prep_w_kernel<<<1024, 256, 0, stream>>>(W2, W2s);
    prep_w_kernel<<<1024, 256, 0, stream>>>(W3, W3s);

    // parallel precompute: Xp = X @ Wx, staged directly in d_out
    xwx_kernel<<<4096, 256, 0, stream>>>(X, Wxs, out);

    // persistent scan: 8 independent 16-row stripes, whole recurrence
    // in-kernel with LDS-resident state and workgroup split-barriers
    scan_kernel<<<Bn / 16, 1024, 0, stream>>>(S0, Whs, W1s, W2s, W3s, out);

    // final_state = outputs[T-1]
    hipMemcpyAsync(out + (size_t)Tn * Bn * Cn,
                   out + (size_t)(Tn - 1) * Bn * Cn,
                   (size_t)Bn * Cn * sizeof(float),
                   hipMemcpyDeviceToDevice, stream);
}
